// CrystalGraphConvNet_24867860644306
// MI455X (gfx1250) — matrix operand order repaired and checked by reading, not compiled, not run on hardware
//
#include <hip/hip_runtime.h>
#include <hip/hip_bf16.h>
#include <stdint.h>

// ---------------- problem constants (match reference) ----------------
#define CC     64      // atom_fea_len
#define MM     12      // max neighbors
#define ORIGF  92      // orig_atom_fea_len
#define NBRF   41      // nbr_fea_len
#define TC     128     // 2*CC gated width
#define KPAD   192     // 2*CC + NBRF = 169 padded to multiple of 32
#define NCONV  3
#define HF     128
#define GFW    16
#define EPSBN  1e-5f

// ---------------- conv-GEMM tiling ----------------
#define TATOMS 8                 // atoms per block
#define TROWS  (TATOMS * MM)     // 96 edge rows per block
#define NTHR   192               // 6 wave32
#define SA     200               // LDS A-tile stride in bf16 (16B-aligned rows, bank-conflict free)
#define SSTG   65                // stage stride in f32

typedef __attribute__((ext_vector_type(16))) __bf16 v16bf;
typedef __attribute__((ext_vector_type(8)))  float  v8f;

union FragAB { v16bf v; uint4 q[2]; };

__device__ __forceinline__ unsigned short f2bf(float x) {
  union { float f; unsigned u; } v; v.f = x;
  unsigned r = v.u + 0x7fffu + ((v.u >> 16) & 1u);   // round-to-nearest-even
  return (unsigned short)(r >> 16);
}
__device__ __forceinline__ float softplus_f(float x) {
  return fmaxf(x, 0.f) + __logf(1.f + __expf(-fabsf(x)));
}

// ---------------- embedding: x = atom_fea @ emb_W + emb_b ----------------
__global__ void embed_kernel(const float* __restrict__ atom_fea,
                             const float* __restrict__ W,
                             const float* __restrict__ b,
                             float* __restrict__ x,
                             unsigned short* __restrict__ xb, int N) {
  __shared__ float af[4][ORIGF];
  int a0 = blockIdx.x * 4;
  int tid = threadIdx.x;
  for (int i = tid; i < 4 * ORIGF; i += 256) {
    int s = i / ORIGF, k = i % ORIGF;
    int a = a0 + s;
    af[s][k] = (a < N) ? atom_fea[(size_t)a * ORIGF + k] : 0.f;
  }
  __syncthreads();
  int s = tid >> 6, c = tid & 63;
  int a = a0 + s;
  if (a >= N) return;
  float acc = b[c];
  #pragma unroll 4
  for (int k = 0; k < ORIGF; ++k) acc += af[s][k] * W[k * CC + c];
  x[(size_t)a * CC + c]  = acc;
  xb[(size_t)a * CC + c] = f2bf(acc);
}

// ------------- per-layer prep: W~(n,k) = W(k,n) in bf16, zero stats -------------
__global__ void prep_kernel(const float* __restrict__ W,          // (169,128)
                            unsigned short* __restrict__ Wt,      // (128,192)
                            float* __restrict__ gstat1, float* __restrict__ gstat2) {
  int idx = blockIdx.x * blockDim.x + threadIdx.x;
  int total = TC * KPAD;
  for (int i = idx; i < total; i += gridDim.x * blockDim.x) {
    int n = i / KPAD, k = i % KPAD;
    Wt[(size_t)n * KPAD + k] =
        (k < 2 * CC + NBRF) ? f2bf(W[(size_t)k * TC + n]) : (unsigned short)0;
  }
  if (idx < 2 * TC) gstat1[idx] = 0.f;
  if (idx < 2 * CC) gstat2[idx] = 0.f;
}

// ---------------- conv GEMM passes ----------------
// PASS 1: GEMM, accumulate per-column sum/sumsq (BN1 stats)
// PASS 2: GEMM, BN1 affine, sigmoid*softplus, reduce over M -> summed; BN2 stats
template <int PASS>
__global__ __launch_bounds__(NTHR)
void conv_pass_kernel(const unsigned short* __restrict__ xb,
                      const int* __restrict__ nbr_idx,
                      const float* __restrict__ nbr_fea,
                      const unsigned short* __restrict__ Wt,
                      const float* __restrict__ s1, const float* __restrict__ t1,
                      float* __restrict__ gstat1,
                      float* __restrict__ summed,
                      float* __restrict__ gstat2,
                      int N) {
  __shared__ __align__(16) unsigned short sA[TROWS * SA];  // 38400 B (stage aliases this)
  __shared__ float sScale[2 * TC];                         // pass1: stats; pass2: s1|t1
  __shared__ float sRed[TATOMS * CC];

  const int tid = threadIdx.x;
  const int a0  = blockIdx.x * TATOMS;

  if (PASS == 1) {
    for (int i = tid; i < 2 * TC; i += NTHR) sScale[i] = 0.f;
  } else {
    for (int i = tid; i < TC; i += NTHR) { sScale[i] = s1[i]; sScale[TC + i] = t1[i]; }
  }

  // ---- build A tile: rows = [x_self(64) | x_nbr(64) | nbr_fea(41) | 0-pad] in bf16 ----
  uint4* A4 = (uint4*)sA;
  for (int i = tid; i < TROWS * (SA / 8); i += NTHR) {
    int row = i / (SA / 8), ch = i % (SA / 8);   // 25 chunks of 8 bf16 per row
    int a = a0 + row / MM, j = row % MM;
    uint4 val = make_uint4(0u, 0u, 0u, 0u);
    if (a < N) {
      if (ch < 8) {
        val = *(const uint4*)(xb + (size_t)a * CC + ch * 8);
      } else if (ch < 16) {
        int nb = nbr_idx[(size_t)a * MM + j];
        val = *(const uint4*)(xb + (size_t)nb * CC + (ch - 8) * 8);
      } else if (ch < 22) {
        int f0 = (ch - 16) * 8;
        const float* fp = nbr_fea + ((size_t)a * MM + j) * NBRF;
        union { unsigned short h[8]; uint4 u; } tconv;
        #pragma unroll
        for (int t = 0; t < 8; ++t) {
          int f = f0 + t;
          tconv.h[t] = (f < NBRF) ? f2bf(fp[f]) : (unsigned short)0;
        }
        val = tconv.u;
      }
    }
    A4[row * (SA / 8) + ch] = val;
  }
  __syncthreads();

  // ---- WMMA GEMM: 16-row strip per wave x 128 cols, K = 6 x 32 ----
  const int wv = tid >> 5, lane = tid & 31;
  const int l15 = lane & 15, g = lane >> 4;

  v8f acc[8];
  #pragma unroll
  for (int nb = 0; nb < 8; ++nb)
    #pragma unroll
    for (int k = 0; k < 8; ++k) acc[nb][k] = 0.f;

  const unsigned short* Arow = sA + (wv * 16 + l15) * SA;
  #pragma unroll
  for (int kb = 0; kb < 6; ++kb) {
    FragAB af;
    const int kbase = kb * 32 + g * 8;                 // A: lanes 16-31 offset by 8
    af.q[0] = *(const uint4*)(Arow + kbase);           // K = kbase..kbase+7
    af.q[1] = *(const uint4*)(Arow + kbase + 16);      // K = kbase+16..kbase+23
    #pragma unroll
    for (int nb = 0; nb < 8; ++nb) {
      FragAB bf;
      const unsigned short* Bp =
          Wt + (size_t)(nb * 16 + l15) * KPAD + kb * 32 + g * 8;  // col l15, K same order as A
      bf.q[0] = *(const uint4*)(Bp);
      bf.q[1] = *(const uint4*)(Bp + 16);
      acc[nb] = __builtin_amdgcn_wmma_f32_16x16x32_bf16(
          false, af.v, false, bf.v, (short)0, acc[nb], false, false);
    }
  }

  if (PASS == 1) {
    // per-column sum / sumsq into LDS, then 2 global atomics per column per block
    const bool full = (a0 + TATOMS) <= N;
    #pragma unroll
    for (int nb = 0; nb < 8; ++nb) {
      float s = 0.f, ss = 0.f;
      #pragma unroll
      for (int k = 0; k < 8; ++k) {
        float v = acc[nb][k];
        if (!full) {
          int row = wv * 16 + g * 8 + k;
          if (a0 + row / MM >= N) v = 0.f;
        }
        s += v; ss += v * v;
      }
      int c = nb * 16 + l15;
      atomicAdd(&sScale[c], s);
      atomicAdd(&sScale[TC + c], ss);
    }
    __syncthreads();
    for (int i = tid; i < 2 * TC; i += NTHR) atomicAdd(&gstat1[i], sScale[i]);
  } else {
    __syncthreads();  // all waves done reading sA before stage overwrites it
    float* stage = (float*)sA;   // 96 x 65 f32 aliases A tile
    #pragma unroll
    for (int nb = 0; nb < 4; ++nb) {
      int cf = nb * 16 + l15;
      float sf = sScale[cf],      tf = sScale[TC + cf];
      float sc = sScale[cf + CC], tcv = sScale[TC + cf + CC];
      #pragma unroll
      for (int k = 0; k < 8; ++k) {
        float fv  = acc[nb][k]     * sf + tf;       // filter (cols 0..63)
        float cv  = acc[nb + 4][k] * sc + tcv;      // core   (cols 64..127)
        float sig = 1.f / (1.f + __expf(-fv));
        float act = sig * softplus_f(cv);
        int row = wv * 16 + g * 8 + k;
        stage[row * SSTG + cf] = act;
      }
    }
    __syncthreads();
    // reduce over M=12 neighbors per atom, write summed, collect BN2 partials
    for (int t = tid; t < TATOMS * CC; t += NTHR) {
      int al = t >> 6, c = t & 63;
      int a = a0 + al;
      float sum = 0.f;
      if (a < N) {
        #pragma unroll
        for (int j = 0; j < MM; ++j) sum += stage[(al * MM + j) * SSTG + c];
        summed[(size_t)a * CC + c] = sum;
      }
      sRed[t] = (a < N) ? sum : 0.f;
    }
    __syncthreads();
    if (tid < CC) {
      float s = 0.f, ss = 0.f;
      #pragma unroll
      for (int al = 0; al < TATOMS; ++al) { float v = sRed[al * CC + tid]; s += v; ss += v * v; }
      atomicAdd(&gstat2[tid], s);
      atomicAdd(&gstat2[CC + tid], ss);
    }
  }
}

// ---------------- BN stats -> affine (s, t) ----------------
__global__ void bn_params_kernel(const float* __restrict__ stats, float inv_count,
                                 const float* __restrict__ g, const float* __restrict__ b,
                                 float* __restrict__ s_out, float* __restrict__ t_out, int nc) {
  int c = blockIdx.x * blockDim.x + threadIdx.x;
  if (c >= nc) return;
  float mu  = stats[c] * inv_count;
  float var = stats[nc + c] * inv_count - mu * mu;
  float s   = g[c] * rsqrtf(var + EPSBN);
  s_out[c] = s;
  t_out[c] = b[c] - mu * s;
}

// ---------------- x = softplus(x + bn2(summed)); refresh bf16 copy ----------------
__global__ void update_kernel(float* __restrict__ x, const float* __restrict__ summed,
                              const float* __restrict__ s2, const float* __restrict__ t2,
                              unsigned short* __restrict__ xb, size_t total) {
  size_t i = (size_t)blockIdx.x * blockDim.x + threadIdx.x;
  if (i >= total) return;
  int c = (int)(i & (CC - 1));
  float v = x[i] + summed[i] * s2[c] + t2[c];
  float r = softplus_f(v);
  x[i]  = r;
  xb[i] = f2bf(r);
}

// ---------------- pooling + MLP head ----------------
__global__ void pool_kernel(const float* __restrict__ x, const int* __restrict__ cidx,
                            const float* __restrict__ gfea,
                            const float* __restrict__ W1, const float* __restrict__ b1,
                            const float* __restrict__ Wo, const float* __restrict__ bo,
                            float* __restrict__ out, int N0) {
  __shared__ float sin[CC + GFW];
  __shared__ float red[HF];
  int n0 = blockIdx.x;
  if (n0 >= N0) return;
  int tid = threadIdx.x;
  int a = cidx[n0];
  if (tid < CC)            sin[tid] = softplus_f(x[(size_t)a * CC + tid]);
  else if (tid < CC + GFW) sin[tid] = softplus_f(gfea[(size_t)n0 * GFW + (tid - CC)]);
  __syncthreads();
  float acc = b1[tid];
  #pragma unroll
  for (int k = 0; k < CC + GFW; ++k) acc += sin[k] * W1[k * HF + tid];
  float h = softplus_f(acc);
  red[tid] = h * Wo[tid];
  __syncthreads();
  for (int s = HF / 2; s > 0; s >>= 1) {
    if (tid < s) red[tid] += red[tid + s];
    __syncthreads();
  }
  if (tid == 0) out[n0] = red[0] + bo[0];
}

// ---------------- host driver ----------------
extern "C" void kernel_launch(void* const* d_in, const int* in_sizes, int n_in,
                              void* d_out, int out_size, void* d_ws, size_t ws_size,
                              hipStream_t stream) {
  const float* atom_fea = (const float*)d_in[0];
  const float* nbr_fea  = (const float*)d_in[1];
  const int*   nbr_idx  = (const int*)d_in[2];
  const int*   cidx     = (const int*)d_in[3];
  const float* gfea     = (const float*)d_in[8];
  const float* emb_W    = (const float*)d_in[9];
  const float* emb_b    = (const float*)d_in[10];
  const float* conv_W   = (const float*)d_in[11];
  // d_in[12] = conv_fc_b: cancels inside training-mode BN -> unused
  const float* bn1_g    = (const float*)d_in[13];
  const float* bn1_b    = (const float*)d_in[14];
  const float* bn2_g    = (const float*)d_in[15];
  const float* bn2_b    = (const float*)d_in[16];
  const float* fc1_W    = (const float*)d_in[17];
  const float* fc1_b    = (const float*)d_in[18];
  const float* out_W    = (const float*)d_in[19];
  const float* out_b    = (const float*)d_in[20];

  const int N  = in_sizes[0] / ORIGF;
  const int N0 = in_sizes[3];
  (void)n_in; (void)out_size; (void)ws_size;

  // workspace carve-out (~64 MB)
  char* ws = (char*)d_ws;
  size_t off = 0;
  auto carve = [&](size_t bytes) {
    void* p = ws + off;
    off = (off + bytes + 255) & ~(size_t)255;
    return p;
  };
  float*          x      = (float*)carve((size_t)N * CC * 4);
  float*          summed = (float*)carve((size_t)N * CC * 4);
  unsigned short* xb     = (unsigned short*)carve((size_t)N * CC * 2);
  unsigned short* Wt     = (unsigned short*)carve((size_t)TC * KPAD * 2);
  float*          gstat1 = (float*)carve(2 * TC * 4);
  float*          gstat2 = (float*)carve(2 * CC * 4);
  float*          s1     = (float*)carve(TC * 4);
  float*          t1     = (float*)carve(TC * 4);
  float*          s2     = (float*)carve(CC * 4);
  float*          t2     = (float*)carve(CC * 4);

  embed_kernel<<<(N + 3) / 4, 256, 0, stream>>>(atom_fea, emb_W, emb_b, x, xb, N);

  const int nblk = (N + TATOMS - 1) / TATOMS;
  for (int L = 0; L < NCONV; ++L) {
    const float* WL = conv_W + (size_t)L * (2 * CC + NBRF) * TC;
    prep_kernel<<<96, 256, 0, stream>>>(WL, Wt, gstat1, gstat2);
    conv_pass_kernel<1><<<nblk, NTHR, 0, stream>>>(xb, nbr_idx, nbr_fea, Wt,
                                                   nullptr, nullptr, gstat1,
                                                   nullptr, nullptr, N);
    bn_params_kernel<<<1, TC, 0, stream>>>(gstat1, 1.f / ((float)N * MM),
                                           bn1_g + L * TC, bn1_b + L * TC, s1, t1, TC);
    conv_pass_kernel<2><<<nblk, NTHR, 0, stream>>>(xb, nbr_idx, nbr_fea, Wt,
                                                   s1, t1, nullptr,
                                                   summed, gstat2, N);
    bn_params_kernel<<<1, CC, 0, stream>>>(gstat2, 1.f / (float)N,
                                           bn2_g + L * CC, bn2_b + L * CC, s2, t2, CC);
    update_kernel<<<(int)(((size_t)N * CC + 255) / 256), 256, 0, stream>>>(
        x, summed, s2, t2, xb, (size_t)N * CC);
  }

  pool_kernel<<<N0, HF, 0, stream>>>(x, cidx, gfea, fc1_W, fc1_b, out_W, out_b,
                                     (float*)d_out, N0);
}
